// GCN_89807766159819
// MI455X (gfx1250) — matrix operand order repaired and checked by reading, not compiled, not run on hardware
//
#include <hip/hip_runtime.h>
#include <hip/hip_bf16.h>

// ---------------------------------------------------------------------------
// GCN forward for MI455X (gfx1250, wave32, WMMA).
//   h = relu(spmm(x@W1)+b1); h = relu(spmm(h@Wh)+bh); h = spmm(h@W2)+b2;
//   out = log_softmax(h, axis=1)
// GEMMs: v_wmma_f32_16x16x32_bf16 with weights staged in LDS via
// global_load_async_to_lds_b128 (ASYNCcnt). SpMM: edge-parallel f32 atomics.
// ---------------------------------------------------------------------------

typedef __attribute__((ext_vector_type(16))) __bf16       v16bf;
typedef __attribute__((ext_vector_type(8)))  float        v8f;
typedef __attribute__((ext_vector_type(8)))  unsigned int v8u;

__device__ __forceinline__ unsigned int bf16pk(float a, float b) {
    unsigned int ua = __float_as_uint(a);
    unsigned int ub = __float_as_uint(b);
    ua = (ua + 0x7FFFu + ((ua >> 16) & 1u)) >> 16;   // RNE f32 -> bf16
    ub = (ub + 0x7FFFu + ((ub >> 16) & 1u)) >> 16;
    return (ua & 0xFFFFu) | (ub << 16);
}

// --- Pack f32 weight [K x N] into WMMA B-fragment layout (bf16) -------------
// dword t = ((kt*NT + nt)*32 + lane)*8 + v  holds  B[k][n],B[k+1][n]
// with k = kt*32 + (lane>=16 ? 16 : 0) + 2v,  n = nt*16 + (lane&15).
__global__ void gcn_pack_w_kernel(const float* __restrict__ W,
                                  unsigned int* __restrict__ P,
                                  int K, int N, int NT) {
    int t = blockIdx.x * blockDim.x + threadIdx.x;
    int total = (K >> 5) * NT * 256;
    if (t >= total) return;
    int v    = t & 7;
    int lane = (t >> 3) & 31;
    int frag = t >> 8;
    int nt   = frag % NT;
    int kt   = frag / NT;
    int n = nt * 16 + (lane & 15);
    int k = kt * 32 + ((lane >> 4) << 4) + 2 * v;
    float w0 = (n < N) ? W[(size_t)k * N + n]       : 0.0f;
    float w1 = (n < N) ? W[(size_t)(k + 1) * N + n] : 0.0f;
    P[t] = bf16pk(w0, w1);
}

// --- WMMA GEMM: Out[M x N] = X[M x (KT*32)] @ Wpacked -----------------------
// Whole packed weight staged in LDS (async copy), shared by all 8 waves.
// One wave computes a 16-row tile across all NT column tiles.
template <int KT, int NT>
__global__ void gcn_gemm_wmma_kernel(const float* __restrict__ X,
                                     const unsigned int* __restrict__ Wp,
                                     float* __restrict__ Out,
                                     int M, int N) {
    constexpr int K = KT * 32;
    __shared__ uint4 ldsW[KT * NT * 64];   // KT*NT fragments, 1KB each

    // ---- async copy packed weights (KT*NT*1024 bytes) into LDS ----
    {
        const unsigned ldsBase = (unsigned)(size_t)(void*)ldsW;  // addr[31:0] = LDS offset
        const unsigned long long gBase = (unsigned long long)Wp;
        const int chunks = KT * NT * 64;   // 16B chunks
        for (int i = threadIdx.x; i < chunks; i += blockDim.x) {
            unsigned dst = ldsBase + (unsigned)i * 16u;
            unsigned long long src = gBase + (unsigned long long)i * 16ull;
            asm volatile("global_load_async_to_lds_b128 %0, %1, off"
                         :: "v"(dst), "v"(src) : "memory");
        }
        asm volatile("s_wait_asynccnt 0x0" ::: "memory");
    }
    __syncthreads();

    const int lane = threadIdx.x & 31;
    const int wv   = threadIdx.x >> 5;
    const int tile = blockIdx.x * (blockDim.x >> 5) + wv;   // 16-row tile id
    if (tile * 16 >= M) return;                              // whole-wave uniform
    const int laneLo = lane & 15;
    const int laneHi = lane >> 4;

    const float* xr = X + ((size_t)tile * 16 + laneLo) * K;

    v8f acc[NT];
#pragma unroll
    for (int i = 0; i < NT; ++i) acc[i] = v8f{};

#pragma unroll 1
    for (int kt = 0; kt < KT; ++kt) {
        const int k0 = kt * 32;
        // A fragment: contiguous K order (lane-lo K 0..15, lane-hi K 16..31), matches packed B
        const float4* pa0 = (const float4*)(xr + k0 + laneHi * 16);
        const float4* pa1 = (const float4*)(xr + k0 + laneHi * 16 + 8);
        float4 f0 = pa0[0], f1 = pa0[1];
        float4 f2 = pa1[0], f3 = pa1[1];
        v8u au;
        au[0] = bf16pk(f0.x, f0.y); au[1] = bf16pk(f0.z, f0.w);
        au[2] = bf16pk(f1.x, f1.y); au[3] = bf16pk(f1.z, f1.w);
        au[4] = bf16pk(f2.x, f2.y); au[5] = bf16pk(f2.z, f2.w);
        au[6] = bf16pk(f3.x, f3.y); au[7] = bf16pk(f3.z, f3.w);
        v16bf a = __builtin_bit_cast(v16bf, au);

#pragma unroll
        for (int nt = 0; nt < NT; ++nt) {
            const uint4* pb = &ldsW[((kt * NT + nt) * 32 + lane) * 2];
            uint4 b0 = pb[0], b1 = pb[1];
            v8u bu;
            bu[0] = b0.x; bu[1] = b0.y; bu[2] = b0.z; bu[3] = b0.w;
            bu[4] = b1.x; bu[5] = b1.y; bu[6] = b1.z; bu[7] = b1.w;
            v16bf b = __builtin_bit_cast(v16bf, bu);
            acc[nt] = __builtin_amdgcn_wmma_f32_16x16x32_bf16(
                false, a, false, b, (short)0, acc[nt], false, false);
        }
    }

    // C/D layout: lanes 0-15 -> rows i, lanes 16-31 -> rows 8+i; col = laneLo
    const size_t rb = (size_t)tile * 16 + laneHi * 8;
#pragma unroll
    for (int nt = 0; nt < NT; ++nt) {
        int col = nt * 16 + laneLo;
        if (col < N) {
#pragma unroll
            for (int i = 0; i < 8; ++i)
                Out[(rb + i) * N + col] = acc[nt][i];
        }
    }
}

// --- Zero fill --------------------------------------------------------------
__global__ void gcn_zero_kernel(float* __restrict__ p, size_t n) {
    size_t i = (size_t)blockIdx.x * blockDim.x + threadIdx.x;
    if (i < n) p[i] = 0.0f;
}

// --- SpMM scatter: out[dst] += w * h[src], one wave per edge ----------------
__global__ void gcn_spmm_kernel(const float* __restrict__ h,
                                const int* __restrict__ esrc,
                                const int* __restrict__ edst,
                                const float* __restrict__ ew,
                                float* __restrict__ out,
                                int nEdges, int F) {
    size_t t = (size_t)blockIdx.x * blockDim.x + threadIdx.x;
    int e = (int)(t >> 5);
    int r = (int)(t & 31);
    if (e >= nEdges) return;
    int s = esrc[e];
    int d = edst[e];
    float w = ew[e];
    const float* hs = h + (size_t)s * F;
    float* od = out + (size_t)d * F;
    for (int f = r; f < F; f += 32)
        atomicAdd(&od[f], w * hs[f]);
}

// --- bias + relu ------------------------------------------------------------
__global__ void gcn_bias_relu_kernel(float* __restrict__ buf,
                                     const float* __restrict__ bias,
                                     size_t total, int F, int relu) {
    size_t i = (size_t)blockIdx.x * blockDim.x + threadIdx.x;
    if (i >= total) return;
    float v = buf[i] + bias[(int)(i % F)];
    if (relu) v = fmaxf(v, 0.0f);
    buf[i] = v;
}

// --- bias + log_softmax (one wave per row, 32 < F <= 64) --------------------
__global__ void gcn_bias_logsoftmax_kernel(float* __restrict__ out,
                                           const float* __restrict__ bias,
                                           int M, int F) {
    int row = blockIdx.x * (blockDim.x >> 5) + (threadIdx.x >> 5);
    int r = threadIdx.x & 31;
    if (row >= M) return;
    float* p = out + (size_t)row * F;
    float v0 = p[r] + bias[r];
    bool has1 = (r + 32) < F;
    float v1 = has1 ? p[r + 32] + bias[r + 32] : -3.4e38f;
    float m = fmaxf(v0, v1);
#pragma unroll
    for (int o = 16; o > 0; o >>= 1) m = fmaxf(m, __shfl_xor(m, o));
    float s = expf(v0 - m) + (has1 ? expf(v1 - m) : 0.0f);
#pragma unroll
    for (int o = 16; o > 0; o >>= 1) s += __shfl_xor(s, o);
    float lse = m + logf(s);
    p[r] = v0 - lse;
    if (has1) p[r + 32] = v1 - lse;
}

// ---------------------------------------------------------------------------
extern "C" void kernel_launch(void* const* d_in, const int* in_sizes, int n_in,
                              void* d_out, int out_size, void* d_ws, size_t ws_size,
                              hipStream_t stream) {
    const float* x    = (const float*)d_in[0];
    const int*   esrc = (const int*)  d_in[1];
    const int*   edst = (const int*)  d_in[2];
    const float* ew   = (const float*)d_in[3];
    const float* W1   = (const float*)d_in[4];
    const float* b1   = (const float*)d_in[5];
    const float* Wh   = (const float*)d_in[6];
    const float* bh   = (const float*)d_in[7];
    const float* W2   = (const float*)d_in[8];
    const float* b2   = (const float*)d_in[9];
    float* out = (float*)d_out;

    const int NFEAT = 512, NHID = 128, NCLASS = 40;
    const int NT_H = NHID / 16;              // 8
    const int NT_C = (NCLASS + 15) / 16;     // 3 (padded to 48)
    const int M = in_sizes[0] / NFEAT;
    const int E = in_sizes[1];

    // workspace carve-out
    char* ws = (char*)d_ws;
    size_t off = 0;
    auto carve = [&](size_t bytes) -> void* {
        void* p = ws + off;
        off = (off + bytes + 255) & ~(size_t)255;
        return p;
    };
    float* bufA = (float*)carve((size_t)M * NHID * sizeof(float));
    float* bufB = (float*)carve((size_t)M * NHID * sizeof(float));
    float* bufC = (float*)carve((size_t)M * NCLASS * sizeof(float));
    unsigned int* wp1 = (unsigned int*)carve((size_t)(NFEAT / 32) * NT_H * 256 * 4);
    unsigned int* wph = (unsigned int*)carve((size_t)(NHID  / 32) * NT_H * 256 * 4);
    unsigned int* wp2 = (unsigned int*)carve((size_t)(NHID  / 32) * NT_C * 256 * 4);

    const int B = 256;
    auto blks = [](size_t n, int b) { return (unsigned)((n + b - 1) / b); };

    // pack weights into WMMA B-fragment layout
    gcn_pack_w_kernel<<<blks((size_t)(NFEAT/32)*NT_H*256, B), B, 0, stream>>>(W1, wp1, NFEAT, NHID, NT_H);
    gcn_pack_w_kernel<<<blks((size_t)(NHID /32)*NT_H*256, B), B, 0, stream>>>(Wh, wph, NHID, NHID, NT_H);
    gcn_pack_w_kernel<<<blks((size_t)(NHID /32)*NT_C*256, B), B, 0, stream>>>(W2, wp2, NHID, NCLASS, NT_C);

    const unsigned gemmBlocks = blks((size_t)((M + 15) / 16), 8);  // 8 waves/block

    // layer 1: gemm -> spmm -> bias+relu
    gcn_gemm_wmma_kernel<16, 8><<<gemmBlocks, B, 0, stream>>>(x, wp1, bufA, M, NHID);
    gcn_zero_kernel<<<blks((size_t)M * NHID, B), B, 0, stream>>>(bufB, (size_t)M * NHID);
    gcn_spmm_kernel<<<blks((size_t)E * 32, B), B, 0, stream>>>(bufA, esrc, edst, ew, bufB, E, NHID);
    gcn_bias_relu_kernel<<<blks((size_t)M * NHID, B), B, 0, stream>>>(bufB, b1, (size_t)M * NHID, NHID, 1);

    // layer 2 (hidden)
    gcn_gemm_wmma_kernel<4, 8><<<gemmBlocks, B, 0, stream>>>(bufB, wph, bufA, M, NHID);
    gcn_zero_kernel<<<blks((size_t)M * NHID, B), B, 0, stream>>>(bufB, (size_t)M * NHID);
    gcn_spmm_kernel<<<blks((size_t)E * 32, B), B, 0, stream>>>(bufA, esrc, edst, ew, bufB, E, NHID);
    gcn_bias_relu_kernel<<<blks((size_t)M * NHID, B), B, 0, stream>>>(bufB, bh, (size_t)M * NHID, NHID, 1);

    // layer 3 (classifier) + log_softmax
    gcn_gemm_wmma_kernel<4, 3><<<gemmBlocks, B, 0, stream>>>(bufB, wp2, bufC, M, NCLASS);
    gcn_zero_kernel<<<blks((size_t)M * NCLASS, B), B, 0, stream>>>(out, (size_t)M * NCLASS);
    gcn_spmm_kernel<<<blks((size_t)E * 32, B), B, 0, stream>>>(bufC, esrc, edst, ew, out, E, NCLASS);
    gcn_bias_logsoftmax_kernel<<<blks((size_t)M, 8), B, 0, stream>>>(out, b2, M, NCLASS);

    (void)n_in; (void)out_size; (void)ws_size;
}
